// PcAttnRe_38800734552515
// MI455X (gfx1250) — matrix operand order repaired and checked by reading, not compiled, not run on hardware
//
#include <hip/hip_runtime.h>

typedef __attribute__((ext_vector_type(16))) _Float16 v16h;
typedef __attribute__((ext_vector_type(8)))  float    v8f;

#define BM 64
#define BN 64
#define BKK 32
#define ASTR 36   // A tile LDS row stride (floats): 144B, 16B-aligned rows
#define BSTR 36   // B tile (opB=1, [n][k]) stride
#define BKSTR 68  // B tile (opB=0, [k][n]) stride: 272B, 16B-aligned rows

// ---------------------------------------------------------------------------
// gfx1250 async global->LDS copy (16B per lane, ASYNCcnt-tracked).
// LDS aperture: low 32 bits of a generic LDS pointer are the LDS byte offset.
// ---------------------------------------------------------------------------
__device__ __forceinline__ void async_copy16(const float* g, float* l)
{
    unsigned      lo = (unsigned)(size_t)l;
    unsigned long long ga = (unsigned long long)(size_t)g;
    asm volatile("global_load_async_to_lds_b128 %0, %1, off"
                 :: "v"(lo), "v"(ga) : "memory");
}
__device__ __forceinline__ void async_wait0()
{
    asm volatile("s_wait_asynccnt 0" ::: "memory");
}

// ---------------------------------------------------------------------------
// Generic batched GEMM, C = alpha * A(MxK) * B(KxN), f32 in/out, f16 WMMA core.
// Two-level batch: z1 = z / batch2, z2 = z % batch2; per-operand base offsets.
// opB=0: B stored K-major (ldb = row stride over K). opB=1: B given as (N x K).
// ---------------------------------------------------------------------------
__global__ __launch_bounds__(128)
void k_gemm_wmma(const float* __restrict__ A, const float* __restrict__ Bm,
                 float* __restrict__ C,
                 int M, int N, int K, int lda, int ldb, int ldc,
                 long offA0, long sA1, long sA2,
                 long offB0, long sB1, long sB2,
                 long offC0, long sC1, long sC2,
                 int batch2, int opB, float alpha)
{
    int z  = blockIdx.z;
    int z1 = z / batch2, z2 = z % batch2;
    const float* Ab = A  + offA0 + (long)z1 * sA1 + (long)z2 * sA2;
    const float* Bb = Bm + offB0 + (long)z1 * sB1 + (long)z2 * sB2;
    float*       Cb = C  + offC0 + (long)z1 * sC1 + (long)z2 * sC2;

    __shared__ float Af[BM * ASTR];        // A tile, [m][k], f32
    __shared__ float Bf[BN * BSTR];        // B tile, [n][k] (opB=1) or [k][n] (opB=0; 32*68=2176<2304)

    int tid  = threadIdx.x;
    int wave = tid >> 5;
    int lane = tid & 31;
    int m0 = blockIdx.y * BM;
    int n0 = blockIdx.x * BN;
    int wm = (wave >> 1) * 32;   // wave row within block tile
    int wn = (wave & 1) * 32;    // wave col within block tile
    int lrow = lane & 15;
    int hi   = lane >> 4;

    v8f acc[2][2];
    #pragma unroll
    for (int i = 0; i < 2; ++i)
        #pragma unroll
        for (int j = 0; j < 2; ++j)
            acc[i][j] = (v8f){0.f,0.f,0.f,0.f,0.f,0.f,0.f,0.f};

    const bool mnFull = (m0 + BM <= M) && (n0 + BN <= N);

    for (int kt = 0; kt < K; kt += BKK) {
        bool full = mnFull && (kt + BKK <= K);
        if (full) {
            // ---- async DMA staging: 16B per lane, pure data movement ----
            #pragma unroll
            for (int it = 0; it < 4; ++it) {
                int idx = tid + 128 * it;            // 0..511
                int r   = idx >> 3;                  // 0..63
                int c4  = (idx & 7) << 2;            // 0,4,..,28
                async_copy16(Ab + (long)(m0 + r) * lda + kt + c4, &Af[r * ASTR + c4]);
            }
            if (opB) {
                #pragma unroll
                for (int it = 0; it < 4; ++it) {
                    int idx = tid + 128 * it;
                    int n   = idx >> 3;
                    int c4  = (idx & 7) << 2;
                    async_copy16(Bb + (long)(n0 + n) * ldb + kt + c4, &Bf[n * BSTR + c4]);
                }
            } else {
                #pragma unroll
                for (int it = 0; it < 4; ++it) {
                    int idx = tid + 128 * it;        // 0..511
                    int r   = idx >> 4;              // 0..31 (k rows)
                    int c4  = (idx & 15) << 2;       // 0..60
                    async_copy16(Bb + (long)(kt + r) * ldb + n0 + c4, &Bf[r * BKSTR + c4]);
                }
            }
            if (kt + BKK < K) {
                int gm = m0 + (tid % BM);
                __builtin_prefetch(&Ab[(long)gm * lda + kt + BKK], 0, 1);
            }
            async_wait0();
        } else {
            // ---- guarded scalar staging (tile edges only) ----
            for (int i = tid; i < BM * BKK; i += 128) {
                int r = i >> 5, c = i & 31;
                int gm = m0 + r, gk = kt + c;
                Af[r * ASTR + c] = (gm < M && gk < K) ? Ab[(long)gm * lda + gk] : 0.f;
            }
            if (opB) {
                for (int i = tid; i < BN * BKK; i += 128) {
                    int n = i >> 5, c = i & 31;
                    int gn = n0 + n, gk = kt + c;
                    Bf[n * BSTR + c] = (gn < N && gk < K) ? Bb[(long)gn * ldb + gk] : 0.f;
                }
            } else {
                for (int i = tid; i < BN * BKK; i += 128) {
                    int r = i >> 6, n = i & 63;      // [k][n]
                    int gk = kt + r, gn = n0 + n;
                    Bf[r * BKSTR + n] = (gn < N && gk < K) ? Bb[(long)gk * ldb + gn] : 0.f;
                }
            }
        }
        __syncthreads();

        // ---- build f16 fragments from f32 LDS tiles (contiguous k = 16*hi + s order) ----
        v16h afrag[2], bfrag[2];
        #pragma unroll
        for (int t = 0; t < 2; ++t) {
            const float* ap = &Af[(wm + t * 16 + lrow) * ASTR];
            #pragma unroll
            for (int vg = 0; vg < 8; ++vg) {
                int kb = (hi ? 16 : 0) + 2 * vg;
                afrag[t][2 * vg]     = (_Float16)ap[kb];
                afrag[t][2 * vg + 1] = (_Float16)ap[kb + 1];
            }
            int col = wn + t * 16 + lrow;
            if (opB) {
                const float* bp = &Bf[col * BSTR];
                #pragma unroll
                for (int vg = 0; vg < 8; ++vg) {
                    int kb = (hi ? 16 : 0) + 2 * vg;
                    bfrag[t][2 * vg]     = (_Float16)bp[kb];
                    bfrag[t][2 * vg + 1] = (_Float16)bp[kb + 1];
                }
            } else {
                int base = hi ? 16 : 0;
                #pragma unroll
                for (int vg = 0; vg < 8; ++vg) {
                    bfrag[t][2 * vg]     = (_Float16)Bf[(base + 2 * vg)     * BKSTR + col];
                    bfrag[t][2 * vg + 1] = (_Float16)Bf[(base + 2 * vg + 1) * BKSTR + col];
                }
            }
        }
        #pragma unroll
        for (int i = 0; i < 2; ++i)
            #pragma unroll
            for (int j = 0; j < 2; ++j)
                acc[i][j] = __builtin_amdgcn_wmma_f32_16x16x32_f16(
                    false, afrag[i], false, bfrag[j],
                    (short)0, acc[i][j], false, false);
        __syncthreads();
    }

    // ---- store C per 32-bit C/D layout (uniform fast path) ----
    if (mnFull) {
        #pragma unroll
        for (int i = 0; i < 2; ++i)
            #pragma unroll
            for (int j = 0; j < 2; ++j)
                #pragma unroll
                for (int e = 0; e < 8; ++e) {
                    int gm = m0 + wm + i * 16 + (hi ? 8 + e : e);
                    int gn = n0 + wn + j * 16 + lrow;
                    Cb[(long)gm * ldc + gn] = acc[i][j][e] * alpha;
                }
    } else {
        #pragma unroll
        for (int i = 0; i < 2; ++i)
            #pragma unroll
            for (int j = 0; j < 2; ++j)
                #pragma unroll
                for (int e = 0; e < 8; ++e) {
                    int gm = m0 + wm + i * 16 + (hi ? 8 + e : e);
                    int gn = n0 + wn + j * 16 + lrow;
                    if (gm < M && gn < N)
                        Cb[(long)gm * ldc + gn] = acc[i][j][e] * alpha;
                }
    }
}

// ---------------------------------------------------------------------------
// x[b, 0, :] = cls + pe(0); x[b, 1+n, :] = f0[b, :, n] + pe(1+n)   (C = 256)
// ---------------------------------------------------------------------------
__global__ void k_build_x(const float* __restrict__ f0,
                          const float* __restrict__ cls,
                          float* __restrict__ x)
{
    int row = blockIdx.x;          // 2 * 1025
    int c   = threadIdx.x;         // 256
    int b = row / 1025, s = row % 1025;
    float base = (s == 0) ? cls[c]
                          : f0[((long)b * 256 + c) * 1024 + (s - 1)];
    int pair = c >> 1;
    float ang = (float)s / powf(10000.f, (2.f * (float)pair) / 256.f);
    float pe = (c & 1) ? cosf(ang) : sinf(ang);
    x[(long)row * 256 + c] = base + pe;
}

__global__ void k_layernorm(const float* __restrict__ in,
                            const float* __restrict__ g,
                            const float* __restrict__ bta,
                            float* __restrict__ out, int rows)
{
    int row = blockIdx.x;
    if (row >= rows) return;
    int t = threadIdx.x;           // 256
    float v = in[(long)row * 256 + t];
    __shared__ float red[256];
    red[t] = v; __syncthreads();
    for (int s = 128; s > 0; s >>= 1) { if (t < s) red[t] += red[t + s]; __syncthreads(); }
    float mean = red[0] * (1.f / 256.f); __syncthreads();
    float d = v - mean;
    red[t] = d * d; __syncthreads();
    for (int s = 128; s > 0; s >>= 1) { if (t < s) red[t] += red[t + s]; __syncthreads(); }
    float rstd = rsqrtf(red[0] * (1.f / 256.f) + 1e-5f);
    out[(long)row * 256 + t] = d * rstd * g[t] + bta[t];
}

__global__ void k_softmax(float* __restrict__ s, int T)
{
    long row = blockIdx.x;
    float* p = s + row * (long)T;
    int t = threadIdx.x;           // 256
    __shared__ float red[256];
    float mx = -1e30f;
    for (int i = t; i < T; i += 256) mx = fmaxf(mx, p[i]);
    red[t] = mx; __syncthreads();
    for (int st = 128; st > 0; st >>= 1) { if (t < st) red[t] = fmaxf(red[t], red[t + st]); __syncthreads(); }
    mx = red[0]; __syncthreads();
    float sum = 0.f;
    for (int i = t; i < T; i += 256) { float e = expf(p[i] - mx); p[i] = e; sum += e; }
    red[t] = sum; __syncthreads();
    for (int st = 128; st > 0; st >>= 1) { if (t < st) red[t] += red[t + st]; __syncthreads(); }
    float inv = 1.f / red[0];
    for (int i = t; i < T; i += 256) p[i] *= inv;
}

__global__ void k_add(float* __restrict__ x, const float* __restrict__ y, long n)
{
    long i = (long)blockIdx.x * blockDim.x + threadIdx.x;
    if (i < n) x[i] += y[i];
}

__global__ void k_gelu(float* __restrict__ x, long n)
{
    long i = (long)blockIdx.x * blockDim.x + threadIdx.x;
    if (i < n) {
        float v = x[i];
        float u = 0.7978845608028654f * (v + 0.044715f * v * v * v);
        x[i] = 0.5f * v * (1.f + tanhf(u));
    }
}

// out[b*1024 + t] = mean_h attn[(b*4+h), 0, t+1]
__global__ void k_attn_cls(const float* __restrict__ sc, float* __restrict__ out,
                           int S, int T)
{
    int i = blockIdx.x * blockDim.x + threadIdx.x;
    if (i >= 2 * 1024) return;
    int b = i >> 10, t = i & 1023;
    float s = 0.f;
    for (int h = 0; h < 4; ++h)
        s += sc[((long)(b * 4 + h) * S) * T + (t + 1)];
    out[i] = 0.25f * s;
}

// bitonic argsort (descending, ties -> lower index) of 1024 values per batch;
// emit order[256:768] as re_idx.
__global__ void k_argsort(const float* __restrict__ vals, int* __restrict__ re_idx)
{
    __shared__ float v[1024];
    __shared__ int  id[1024];
    int t = threadIdx.x, b = blockIdx.x;
    v[t] = vals[b * 1024 + t]; id[t] = t;
    __syncthreads();
    for (int k = 2; k <= 1024; k <<= 1)
        for (int j = k >> 1; j > 0; j >>= 1) {
            int ixj = t ^ j;
            if (ixj > t) {
                float va = v[t], vb = v[ixj];
                int ia = id[t], ib = id[ixj];
                bool dirDesc = ((t & k) == 0);
                bool aFirst = (va > vb) || (va == vb && ia < ib);
                if (dirDesc != aFirst) {
                    v[t] = vb; v[ixj] = va; id[t] = ib; id[ixj] = ia;
                }
            }
            __syncthreads();
        }
    if (t >= 256 && t < 768) re_idx[b * 512 + (t - 256)] = id[t];
}

// kNN: one thread per query; keep k smallest squared distances (k <= 16)
__global__ void k_knn(const float* __restrict__ q, const float* __restrict__ r,
                      int* __restrict__ idx, int nq, int nr, int k)
{
    int i = blockIdx.x * blockDim.x + threadIdx.x;
    if (i >= 2 * nq) return;
    int b = i / nq, qi = i % nq;
    const float* qp = q + ((long)b * nq + qi) * 3;
    float qx = qp[0], qy = qp[1], qz = qp[2];
    float bd[16]; int bi[16];
    for (int t = 0; t < k; ++t) { bd[t] = 1e30f; bi[t] = 0; }
    const float* rb = r + (long)b * nr * 3;
    for (int m = 0; m < nr; ++m) {
        float dx = qx - rb[m * 3], dy = qy - rb[m * 3 + 1], dz = qz - rb[m * 3 + 2];
        float d = dx * dx + dy * dy + dz * dz;
        if (d < bd[k - 1]) {
            int p = k - 1;
            while (p > 0 && bd[p - 1] > d) { bd[p] = bd[p - 1]; bi[p] = bi[p - 1]; --p; }
            bd[p] = d; bi[p] = m;
        }
    }
    for (int t = 0; t < k; ++t) idx[(long)i * k + t] = bi[t];
}

// dst[(b*nIdx + i)*F + f] = src[((long)b*srcRows + rowOff + idx[b*nIdx+i])*F + f]
__global__ void k_gather(const float* __restrict__ src, const int* __restrict__ idx,
                         float* __restrict__ dst, long nIdx, int srcRows, int rowOff, int F)
{
    long tot = 2L * nIdx * F;
    long i = (long)blockIdx.x * blockDim.x + threadIdx.x;
    if (i >= tot) return;
    int  f    = (int)(i % F);
    long rest = i / F;
    long ii   = rest % nIdx;
    long b    = rest / nIdx;
    int row = idx[b * nIdx + ii];
    dst[i] = src[((long)b * srcRows + rowOff + row) * F + f];
}

// dst[b, i, f] = src[b, idx[b,i], f]   (ints)
__global__ void k_gather_i(const int* __restrict__ src, const int* __restrict__ idx,
                           int* __restrict__ dst, long nIdx, int srcRows, int F)
{
    long tot = 2L * nIdx * F;
    long i = (long)blockIdx.x * blockDim.x + threadIdx.x;
    if (i >= tot) return;
    int  f    = (int)(i % F);
    long rest = i / F;
    long ii   = rest % nIdx;
    long b    = rest / nIdx;
    int row = idx[b * nIdx + ii];
    dst[i] = src[((long)b * srcRows + row) * F + f];
}

// dst[(b*nIdx+i)*256 + c] = f1[b, c, idx[b,i]]  (channel-major source)
__global__ void k_gather_cm(const float* __restrict__ f1, const int* __restrict__ idx,
                            float* __restrict__ dst, long nIdx, int Nsrc)
{
    long rowi = blockIdx.x;            // 2 * nIdx
    int c = threadIdx.x;               // 256
    long b = rowi / nIdx, i = rowi % nIdx;
    int n = idx[b * nIdx + i];
    dst[rowi * 256 + c] = f1[((long)b * 256 + c) * (long)Nsrc + n];
}

// x[r, c] += coords[r, :3] @ pos[:3, c]
__global__ void k_add_pos(float* __restrict__ x, const float* __restrict__ coords,
                          const float* __restrict__ pos)
{
    long r = blockIdx.x;
    int  c = threadIdx.x;              // 256
    float cx = coords[r * 3], cy = coords[r * 3 + 1], cz = coords[r * 3 + 2];
    x[r * 256 + c] += cx * pos[c] + cy * pos[256 + c] + cz * pos[512 + c];
}

__global__ void k_aggr(const float* __restrict__ x, float* __restrict__ out)
{
    int b = blockIdx.x;
    out[b * 256 + threadIdx.x] = x[(long)b * 1025 * 256 + threadIdx.x];
}

// ---------------------------------------------------------------------------
// Host orchestration
// ---------------------------------------------------------------------------
struct Blk {
    const float *ln1s, *ln1b, *wq, *wkv, *wo, *ln2s, *ln2b, *w1, *w2;
    const float *pos, *lnks, *lnkb;
};

static inline void gemm(hipStream_t st, const float* A, const float* B, float* C,
                        int M, int N, int K, int lda, int ldb, int ldc,
                        long oA, long sA1, long sA2,
                        long oB, long sB1, long sB2,
                        long oC, long sC1, long sC2,
                        int batch1, int batch2, int opB, float alpha)
{
    dim3 g((N + BN - 1) / BN, (M + BM - 1) / BM, batch1 * batch2);
    k_gemm_wmma<<<g, 128, 0, st>>>(A, B, C, M, N, K, lda, ldb, ldc,
                                   oA, sA1, sA2, oB, sB1, sB2, oC, sC1, sC2,
                                   batch2, opB, alpha);
}

extern "C" void kernel_launch(void* const* d_in, const int* in_sizes, int n_in,
                              void* d_out, int out_size, void* d_ws, size_t ws_size,
                              hipStream_t stream)
{
    const float* f0  = (const float*)d_in[0];   // (2,256,1024)
    const float* f1  = (const float*)d_in[1];   // (2,256,4096)
    const float* c0  = (const float*)d_in[2];   // (2,1024,3)
    const float* c1  = (const float*)d_in[3];   // (2,4096,3)
    const float* cls = (const float*)d_in[4];   // (1,1,256)

    auto getBlk = [&](int base, bool pos, bool cross) {
        Blk b{};
        b.ln1s = (const float*)d_in[base + 0]; b.ln1b = (const float*)d_in[base + 1];
        b.wq   = (const float*)d_in[base + 2]; b.wkv  = (const float*)d_in[base + 3];
        b.wo   = (const float*)d_in[base + 4]; b.ln2s = (const float*)d_in[base + 5];
        b.ln2b = (const float*)d_in[base + 6]; b.w1   = (const float*)d_in[base + 7];
        b.w2   = (const float*)d_in[base + 8];
        int e = base + 9;
        if (pos)  { b.pos  = (const float*)d_in[e++]; }
        if (cross){ b.lnks = (const float*)d_in[e++]; b.lnkb = (const float*)d_in[e++]; }
        return b;
    };
    Blk fs0 = getBlk(5, false, false);
    Blk fs1 = getBlk(14, false, false);
    Blk sa1 = getBlk(23, true, false);
    Blk sa2 = getBlk(33, true, false);
    Blk ca  = getBlk(43, false, true);

    // ---------------- workspace layout (floats, then ints) ----------------
    float* wf = (float*)d_ws;
    size_t o = 0;
    auto A_ = [&](size_t n) { float* p = wf + o; o += n; return p; };
    float* x       = A_((size_t)2 * 1025 * 256);
    float* h       = A_((size_t)2 * 1025 * 256);
    float* q       = A_((size_t)2 * 1025 * 256);
    float* kv      = A_((size_t)2 * 1025 * 512);
    float* scores  = A_((size_t)8 * 1025 * 1025);
    float* oh      = A_((size_t)2 * 1025 * 256);
    float* mlp     = A_((size_t)2 * 1025 * 1024);
    float* acls    = A_((size_t)2 * 1024);
    float* reco    = A_((size_t)2 * 512 * 3);
    float* nxreco  = A_((size_t)2 * 4096 * 3);
    float* reco_in = A_((size_t)1024 * 16 * 3);
    float* nxco_in = A_((size_t)1024 * 128 * 3);
    float* ch_h    = A_((size_t)16384 * 256);
    float* ch_h2   = A_((size_t)16384 * 256);
    float* ch_q    = A_((size_t)16384 * 256);
    float* ch_kv   = A_((size_t)16384 * 512);
    float* ch_sc   = A_((size_t)128 * 4 * 128 * 128);
    float* ch_oh   = A_((size_t)16384 * 256);
    float* ch_mlp  = A_((size_t)16384 * 1024);
    int* wi = (int*)(wf + o);
    size_t oi = 0;
    auto I_ = [&](size_t n) { int* p = wi + oi; oi += n; return p; };
    int* re_idx   = I_((size_t)2 * 512);
    int* idx_c2f  = I_((size_t)2 * 1024 * 8);
    int* idx_same = I_((size_t)2 * 512 * 16);
    int* nxt_idx  = I_((size_t)2 * 512 * 8);
    int* idx_nxt  = I_((size_t)2 * 4096 * 16);

    float* out      = (float*)d_out;
    float* aggr_out = out;                                   // (2, 256)
    float* re_feat  = out + 512;                             // (1024, 16, 256)
    float* nxt      = out + 512 + (size_t)1024 * 16 * 256;   // (1024, 128, 256)

    const long SAT  = 1025L * 1025;   // scores per (b,h), first_sa
    const long ROW  = 256;

    // ===================== Stage A: first_sa =====================
    k_build_x<<<2 * 1025, 256, 0, stream>>>(f0, cls, x);
    const Blk* fsa[2] = { &fs0, &fs1 };
    for (int bi = 0; bi < 2; ++bi) {
        const Blk& p = *fsa[bi];
        k_layernorm<<<2050, 256, 0, stream>>>(x, p.ln1s, p.ln1b, h, 2050);
        gemm(stream, h, p.wq,  q,  2050, 256, 256, 256, 256, 256,
             0,0,0, 0,0,0, 0,0,0, 1,1,0, 1.f);
        gemm(stream, h, p.wkv, kv, 2050, 512, 256, 256, 512, 512,
             0,0,0, 0,0,0, 0,0,0, 1,1,0, 1.f);
        // scores = (q @ k^T) / 8
        gemm(stream, q, kv, scores, 1025, 1025, 64, 256, 512, 1025,
             0, 1025L * ROW, 64,
             0, 1025L * 512, 64,
             0, 4 * SAT, SAT,
             2, 4, 1, 0.125f);
        k_softmax<<<8 * 1025, 256, 0, stream>>>(scores, 1025);
        if (bi == 1)
            k_attn_cls<<<(2048 + 255) / 256, 256, 0, stream>>>(scores, acls, 1025, 1025);
        // o_heads = probs @ v
        gemm(stream, scores, kv, oh, 1025, 64, 1025, 1025, 512, 256,
             0, 4 * SAT, SAT,
             256, 1025L * 512, 64,
             0, 1025L * ROW, 64,
             2, 4, 0, 1.f);
        gemm(stream, oh, p.wo, h, 2050, 256, 256, 256, 256, 256,
             0,0,0, 0,0,0, 0,0,0, 1,1,0, 1.f);
        k_add<<<(2050 * 256 + 255) / 256, 256, 0, stream>>>(x, h, 2050L * 256);
        k_layernorm<<<2050, 256, 0, stream>>>(x, p.ln2s, p.ln2b, h, 2050);
        gemm(stream, h, p.w1, mlp, 2050, 1024, 256, 256, 1024, 1024,
             0,0,0, 0,0,0, 0,0,0, 1,1,0, 1.f);
        k_gelu<<<(2050 * 1024 + 255) / 256, 256, 0, stream>>>(mlp, 2050L * 1024);
        gemm(stream, mlp, p.w2, h, 2050, 256, 1024, 1024, 256, 256,
             0,0,0, 0,0,0, 0,0,0, 1,1,0, 1.f);
        k_add<<<(2050 * 256 + 255) / 256, 256, 0, stream>>>(x, h, 2050L * 256);
    }
    k_aggr<<<2, 256, 0, stream>>>(x, aggr_out);
    k_argsort<<<2, 1024, 0, stream>>>(acls, re_idx);

    // ===================== Stage B: kNN + gathers =====================
    k_knn<<<(2 * 1024 + 255) / 256, 256, 0, stream>>>(c0, c1, idx_c2f, 1024, 4096, 8);
    k_gather<<<(2 * 512 * 3 + 255) / 256, 256, 0, stream>>>(c0, re_idx, reco, 512, 1024, 0, 3);
    k_knn<<<(2 * 512 + 255) / 256, 256, 0, stream>>>(reco, c0, idx_same, 512, 1024, 16);
    k_gather<<<(2L * 8192 * 256 + 255) / 256, 256, 0, stream>>>(x, idx_same, re_feat, 8192, 1025, 1, 256);
    k_gather<<<(2 * 8192 * 3 + 255) / 256, 256, 0, stream>>>(c0, idx_same, reco_in, 8192, 1024, 0, 3);
    k_gather_i<<<(2 * 512 * 8 + 255) / 256, 256, 0, stream>>>(idx_c2f, re_idx, nxt_idx, 512, 1024, 8);
    k_gather<<<(2 * 4096 * 3 + 255) / 256, 256, 0, stream>>>(c1, nxt_idx, nxreco, 4096, 4096, 0, 3);
    k_knn<<<(2 * 4096 + 255) / 256, 256, 0, stream>>>(nxreco, c1, idx_nxt, 4096, 4096, 16);
    k_gather<<<(2 * 65536 * 3 + 255) / 256, 256, 0, stream>>>(c1, idx_nxt, nxco_in, 65536, 4096, 0, 3);
    k_gather_cm<<<2 * 65536, 256, 0, stream>>>(f1, idx_nxt, nxt, 65536, 4096);

    // ===================== Stage C: sa1 on nxt (chunked, 8 x 128 seqs) ====
    k_add_pos<<<131072, 256, 0, stream>>>(nxt, nxco_in, sa1.pos);
    for (int cidx = 0; cidx < 8; ++cidx) {
        float* xb = nxt + (size_t)cidx * 16384 * 256;
        k_layernorm<<<16384, 256, 0, stream>>>(xb, sa1.ln1s, sa1.ln1b, ch_h, 16384);
        gemm(stream, ch_h, sa1.wq,  ch_q,  16384, 256, 256, 256, 256, 256,
             0,0,0, 0,0,0, 0,0,0, 1,1,0, 1.f);
        gemm(stream, ch_h, sa1.wkv, ch_kv, 16384, 512, 256, 256, 512, 512,
             0,0,0, 0,0,0, 0,0,0, 1,1,0, 1.f);
        gemm(stream, ch_q, ch_kv, ch_sc, 128, 128, 64, 256, 512, 128,
             0, 128L * 256, 64,
             0, 128L * 512, 64,
             0, 4L * 128 * 128, 128L * 128,
             128, 4, 1, 0.125f);
        k_softmax<<<512 * 128, 256, 0, stream>>>(ch_sc, 128);
        gemm(stream, ch_sc, ch_kv, ch_oh, 128, 64, 128, 128, 512, 256,
             0, 4L * 128 * 128, 128L * 128,
             256, 128L * 512, 64,
             0, 128L * 256, 64,
             128, 4, 0, 1.f);
        gemm(stream, ch_oh, sa1.wo, ch_h, 16384, 256, 256, 256, 256, 256,
             0,0,0, 0,0,0, 0,0,0, 1,1,0, 1.f);
        k_add<<<(16384 * 256 + 255) / 256, 256, 0, stream>>>(xb, ch_h, 16384L * 256);
        k_layernorm<<<16384, 256, 0, stream>>>(xb, sa1.ln2s, sa1.ln2b, ch_h, 16384);
        gemm(stream, ch_h, sa1.w1, ch_mlp, 16384, 1024, 256, 256, 1024, 1024,
             0,0,0, 0,0,0, 0,0,0, 1,1,0, 1.f);
        k_gelu<<<(16384 * 1024 + 255) / 256, 256, 0, stream>>>(ch_mlp, 16384L * 1024);
        gemm(stream, ch_mlp, sa1.w2, ch_h, 16384, 256, 1024, 1024, 256, 256,
             0,0,0, 0,0,0, 0,0,0, 1,1,0, 1.f);
        k_add<<<(16384 * 256 + 255) / 256, 256, 0, stream>>>(xb, ch_h, 16384L * 256);
    }

    // ===================== Stage D: sa2 on re_feat (1024 x 16) ===========
    k_add_pos<<<16384, 256, 0, stream>>>(re_feat, reco_in, sa2.pos);
    {
        k_layernorm<<<16384, 256, 0, stream>>>(re_feat, sa2.ln1s, sa2.ln1b, ch_h, 16384);
        gemm(stream, ch_h, sa2.wq,  ch_q,  16384, 256, 256, 256, 256, 256,
             0,0,0, 0,0,0, 0,0,0, 1,1,0, 1.f);
        gemm(stream, ch_h, sa2.wkv, ch_kv, 16384, 512, 256, 256, 512, 512,
             0,0,0, 0,0,0, 0,0,0, 1,1,0, 1.f);
        gemm(stream, ch_q, ch_kv, ch_sc, 16, 16, 64, 256, 512, 16,
             0, 16L * 256, 64,
             0, 16L * 512, 64,
             0, 4L * 16 * 16, 16L * 16,
             1024, 4, 1, 0.125f);
        k_softmax<<<4096 * 16, 256, 0, stream>>>(ch_sc, 16);
        gemm(stream, ch_sc, ch_kv, ch_oh, 16, 64, 16, 16, 512, 256,
             0, 4L * 16 * 16, 16L * 16,
             256, 16L * 512, 64,
             0, 16L * 256, 64,
             1024, 4, 0, 1.f);
        gemm(stream, ch_oh, sa2.wo, ch_h, 16384, 256, 256, 256, 256, 256,
             0,0,0, 0,0,0, 0,0,0, 1,1,0, 1.f);
        k_add<<<(16384 * 256 + 255) / 256, 256, 0, stream>>>(re_feat, ch_h, 16384L * 256);
        k_layernorm<<<16384, 256, 0, stream>>>(re_feat, sa2.ln2s, sa2.ln2b, ch_h, 16384);
        gemm(stream, ch_h, sa2.w1, ch_mlp, 16384, 1024, 256, 256, 1024, 1024,
             0,0,0, 0,0,0, 0,0,0, 1,1,0, 1.f);
        k_gelu<<<(16384 * 1024 + 255) / 256, 256, 0, stream>>>(ch_mlp, 16384L * 1024);
        gemm(stream, ch_mlp, sa2.w2, ch_h, 16384, 256, 1024, 1024, 256, 256,
             0,0,0, 0,0,0, 0,0,0, 1,1,0, 1.f);
        k_add<<<(16384 * 256 + 255) / 256, 256, 0, stream>>>(re_feat, ch_h, 16384L * 256);
    }

    // ===================== Stage E: cross-attn (chunked, 8 x 128 seqs) ====
    for (int cidx = 0; cidx < 8; ++cidx) {
        float* xb  = re_feat + (size_t)cidx * 2048 * 256;   // 128 seqs x 16 toks
        float* kvb = nxt     + (size_t)cidx * 16384 * 256;  // 128 seqs x 128 toks
        k_layernorm<<<2048, 256, 0, stream>>>(xb, ca.ln1s, ca.ln1b, ch_h2, 2048);
        k_layernorm<<<16384, 256, 0, stream>>>(kvb, ca.lnks, ca.lnkb, ch_h, 16384);
        gemm(stream, ch_h2, ca.wq,  ch_q,  2048, 256, 256, 256, 256, 256,
             0,0,0, 0,0,0, 0,0,0, 1,1,0, 1.f);
        gemm(stream, ch_h,  ca.wkv, ch_kv, 16384, 512, 256, 256, 512, 512,
             0,0,0, 0,0,0, 0,0,0, 1,1,0, 1.f);
        gemm(stream, ch_q, ch_kv, ch_sc, 16, 128, 64, 256, 512, 128,
             0, 16L * 256, 64,
             0, 128L * 512, 64,
             0, 4L * 16 * 128, 16L * 128,
             128, 4, 1, 0.125f);
        k_softmax<<<512 * 16, 256, 0, stream>>>(ch_sc, 128);
        gemm(stream, ch_sc, ch_kv, ch_oh, 16, 64, 128, 128, 512, 256,
             0, 4L * 16 * 128, 16L * 128,
             256, 128L * 512, 64,
             0, 16L * 256, 64,
             128, 4, 0, 1.f);
        gemm(stream, ch_oh, ca.wo, ch_h2, 2048, 256, 256, 256, 256, 256,
             0,0,0, 0,0,0, 0,0,0, 1,1,0, 1.f);
        k_add<<<(2048 * 256 + 255) / 256, 256, 0, stream>>>(xb, ch_h2, 2048L * 256);
        k_layernorm<<<2048, 256, 0, stream>>>(xb, ca.ln2s, ca.ln2b, ch_h2, 2048);
        gemm(stream, ch_h2, ca.w1, ch_mlp, 2048, 1024, 256, 256, 1024, 1024,
             0,0,0, 0,0,0, 0,0,0, 1,1,0, 1.f);
        k_gelu<<<(2048 * 1024 + 255) / 256, 256, 0, stream>>>(ch_mlp, 2048L * 1024);
        gemm(stream, ch_mlp, ca.w2, ch_h2, 2048, 256, 1024, 1024, 256, 256,
             0,0,0, 0,0,0, 0,0,0, 1,1,0, 1.f);
        k_add<<<(2048 * 256 + 255) / 256, 256, 0, stream>>>(xb, ch_h2, 2048L * 256);
    }
}
